// Router_83846351553050
// MI455X (gfx1250) — matrix operand order repaired and checked by reading, not compiled, not run on hardware
//
#include <hip/hip_runtime.h>
#include <hip/hip_bf16.h>

// ---------------------------------------------------------------------------
// Problem constants (from reference)
// ---------------------------------------------------------------------------
constexpr int BATCH = 8;
constexpr int SEQ   = 2048;
constexpr int DIM   = 4096;
constexpr int NEXP  = 64;
constexpr int CAP   = 40;                 // EXPERT_CAPACITY
constexpr int MTOK  = BATCH * SEQ;        // 16384 tokens
constexpr int KSTEPS = DIM / 32;          // 128 WMMA K-steps

typedef __attribute__((ext_vector_type(16))) __bf16 v16bf;
typedef __attribute__((ext_vector_type(2)))  __bf16 v2bf;
typedef __attribute__((ext_vector_type(8)))  float  v8f;

// Packed f32 -> bf16 conversion (v_cvt_pk_bf16_f32, confirmed in round-2 asm).
static __device__ __forceinline__ v2bf pk_bf16(float lo, float hi) {
#if __has_builtin(__builtin_amdgcn_cvt_pk_bf16_f32)
    return __builtin_amdgcn_cvt_pk_bf16_f32(lo, hi);
#else
    v2bf r; r[0] = (__bf16)lo; r[1] = (__bf16)hi; return r;
#endif
}

static __device__ __forceinline__ __bf16 f2bf(float f) {
    unsigned u = __builtin_bit_cast(unsigned, f);
    unsigned r = (u + 0x7FFFu + ((u >> 16) & 1u)) >> 16;   // round-nearest-even
    return __builtin_bit_cast(__bf16, (unsigned short)r);
}

// Load 16 f32, p[0..7] and p[16..23], with p (= row + k0 + half*8 pattern handled by
// caller) and pack into the ISA 16-bit A-matrix lane layout (K 0..7,16..23 /
// 8..15,24..31 split handled via the caller's +half*8 base).
static __device__ __forceinline__ v16bf load_a_bf16(const float* __restrict__ p) {
    const float4* pa = (const float4*)p;
    const float4* pb = (const float4*)(p + 16);
    float4 a0 = pa[0], a1 = pa[1];
    float4 a2 = pb[0], a3 = pb[1];
    union { v16bf v; v2bf h[8]; } au;
    au.h[0] = pk_bf16(a0.x, a0.y);
    au.h[1] = pk_bf16(a0.z, a0.w);
    au.h[2] = pk_bf16(a1.x, a1.y);
    au.h[3] = pk_bf16(a1.z, a1.w);
    au.h[4] = pk_bf16(a2.x, a2.y);
    au.h[5] = pk_bf16(a2.z, a2.w);
    au.h[6] = pk_bf16(a3.x, a3.y);
    au.h[7] = pk_bf16(a3.z, a3.w);
    return au.v;
}

// ---------------------------------------------------------------------------
// Kernel 0: swizzle W (f32 [D,E] row-major) into WMMA B-matrix lane layout, bf16.
// For kstep t (K block of 32), tile nt (N block of 16), lane (0..31):
//   16 contiguous bf16 = W[k0 + (lane>=16?8:0) + (j&7) + (j>=8?16:0)][nt*16 + (lane&15)], j=0..15
// ---------------------------------------------------------------------------
__global__ void pack_w(const float* __restrict__ W, __bf16* __restrict__ Wb) {
    int i = blockIdx.x * blockDim.x + threadIdx.x;     // 0 .. D*E-1 = 262143
    if (i >= DIM * NEXP) return;
    int j    =  i        & 15;
    int lane = (i >> 4)  & 31;
    int nt   = (i >> 9)  & 3;
    int t    =  i >> 11;
    int n = nt * 16 + (lane & 15);
    int k = t * 32 + ((lane >> 4) << 3) + (j & 7) + ((j >> 3) << 4);
    Wb[i] = f2bf(W[k * NEXP + n]);
}

// ---------------------------------------------------------------------------
// Kernel 1: fused GEMM (bf16 WMMA) + bias + softmax + top1/argmax.
// ONE wave32 per workgroup; each wave owns 32 tokens (two 16-row M-tiles),
// so each B tile register feeds 2 WMMAs (halves L2 traffic on W) and the
// softmax phase uses all 32 lanes. Grid = 512 blocks for device fill.
// ---------------------------------------------------------------------------
__global__ __launch_bounds__(32) void router_gemm_softmax(
    const float*  __restrict__ X,      // [MTOK, DIM] f32
    const __bf16* __restrict__ Wb,     // swizzled, see pack_w
    const float*  __restrict__ bias,   // [NEXP]
    float* __restrict__ probs_out,     // [MTOK, NEXP]
    float* __restrict__ top1_out,      // [MTOK]
    int*   __restrict__ eidx_out)      // [MTOK]
{
    __shared__ float logit_lds[32][65];      // 65: bank-conflict padding

    const int lane = threadIdx.x & 31;
    const int half = lane >> 4;              // 0 | 1
    const int l    = lane & 15;
    const int m0   = blockIdx.x * 32;

    const float* xrow0 = X + (size_t)(m0 + l) * DIM + half * 8;
    const float* xrow1 = X + (size_t)(m0 + 16 + l) * DIM + half * 8;

    v8f a00 = {}, a01 = {}, a02 = {}, a03 = {};   // M-tile 0, N-tiles 0..3
    v8f a10 = {}, a11 = {}, a12 = {}, a13 = {};   // M-tile 1, N-tiles 0..3

    for (int t = 0; t < KSTEPS; ++t) {
        const int k0 = t * 32;
        if (t + 8 < KSTEPS) {                      // ~1KB ahead per row
            __builtin_prefetch(xrow0 + k0 + 256, 0, 3);
            __builtin_prefetch(xrow1 + k0 + 256, 0, 3);
        }

        const v16bf A0 = load_a_bf16(xrow0 + k0);
        const v16bf A1 = load_a_bf16(xrow1 + k0);

        // ---- B operands: one coalesced 32B load per lane per tile ----
        const v16bf* wp = (const v16bf*)(Wb + ((size_t)(t * 4) * 32 + lane) * 16);
        v16bf B0 = wp[0];
        v16bf B1 = wp[32];     // +32*16 bf16 per N-tile
        v16bf B2 = wp[64];
        v16bf B3 = wp[96];

        a00 = __builtin_amdgcn_wmma_f32_16x16x32_bf16(false, A0, false, B0, (short)0, a00, false, false);
        a10 = __builtin_amdgcn_wmma_f32_16x16x32_bf16(false, A1, false, B0, (short)0, a10, false, false);
        a01 = __builtin_amdgcn_wmma_f32_16x16x32_bf16(false, A0, false, B1, (short)0, a01, false, false);
        a11 = __builtin_amdgcn_wmma_f32_16x16x32_bf16(false, A1, false, B1, (short)0, a11, false, false);
        a02 = __builtin_amdgcn_wmma_f32_16x16x32_bf16(false, A0, false, B2, (short)0, a02, false, false);
        a12 = __builtin_amdgcn_wmma_f32_16x16x32_bf16(false, A1, false, B2, (short)0, a12, false, false);
        a03 = __builtin_amdgcn_wmma_f32_16x16x32_bf16(false, A0, false, B3, (short)0, a03, false, false);
        a13 = __builtin_amdgcn_wmma_f32_16x16x32_bf16(false, A1, false, B3, (short)0, a13, false, false);
    }

    // C layout: VGPR i, lanes 0-15 -> (M=i, N=lane); lanes 16-31 -> (M=8+i, N=lane-16)
    const int rbase = half * 8;
    #pragma unroll
    for (int i = 0; i < 8; ++i) {
        logit_lds[rbase + i][ 0 + l] = a00[i];
        logit_lds[rbase + i][16 + l] = a01[i];
        logit_lds[rbase + i][32 + l] = a02[i];
        logit_lds[rbase + i][48 + l] = a03[i];
        logit_lds[16 + rbase + i][ 0 + l] = a10[i];
        logit_lds[16 + rbase + i][16 + l] = a11[i];
        logit_lds[16 + rbase + i][32 + l] = a12[i];
        logit_lds[16 + rbase + i][48 + l] = a13[i];
    }
    __syncthreads();   // single-wave workgroup: barrier is a HW NOP + LDS fence

    // ---- softmax over 64 experts: one token per lane (all 32 lanes busy) ----
    {
        const int m = m0 + lane;
        float vmax = -3.0e38f; int amax = 0;
        #pragma unroll
        for (int e = 0; e < NEXP; ++e) {
            float x = logit_lds[lane][e] + bias[e];
            if (x > vmax) { vmax = x; amax = e; }
        }
        float sum = 0.f;
        #pragma unroll
        for (int e = 0; e < NEXP; ++e)
            sum += __expf(logit_lds[lane][e] + bias[e] - vmax);
        const float inv = 1.f / sum;
        float* prow = probs_out + (size_t)m * NEXP;
        #pragma unroll
        for (int e = 0; e < NEXP; ++e)
            prow[e] = __expf(logit_lds[lane][e] + bias[e] - vmax) * inv;
        top1_out[m] = inv;          // exp(vmax - vmax) / sum
        eidx_out[m] = amax;
    }
}

// ---------------------------------------------------------------------------
// Kernel 2a: capacity cumsum. One wave32 per (b, e) pair; ballot prefix-scan.
// ---------------------------------------------------------------------------
__global__ __launch_bounds__(256) void capacity_scan(
    const int* __restrict__ eidx,     // [MTOK]
    int* __restrict__ keep,           // [MTOK]
    int* __restrict__ kept_cnt,       // [BATCH*NEXP]
    int* __restrict__ drop_cnt)       // [BATCH*NEXP]
{
    const int lane = threadIdx.x & 31;
    const int w    = threadIdx.x >> 5;
    const int p    = blockIdx.x * 8 + w;       // 0..511
    const int b    = p >> 6;
    const int e    = p & 63;
    const int* idx = eidx + b * SEQ;

    int running = 0;
    for (int s0 = 0; s0 < SEQ; s0 += 32) {
        const int idv = idx[s0 + lane];
        const bool match = (idv == e);
        const unsigned mask = __builtin_amdgcn_ballot_w32(match);
        const int prio = running + __popc(mask & ((1u << lane) - 1u)) + 1;
        if (match) keep[b * SEQ + s0 + lane] = (prio <= CAP) ? 1 : 0;
        running += __popc(mask);
    }
    if (lane == 0) {
        kept_cnt[p] = running < CAP ? running : CAP;
        drop_cnt[p] = running > CAP ? running - CAP : 0;
    }
}

// ---------------------------------------------------------------------------
// Kernel 2b: coalesced one-hot expansion into d_out (float 0/1).
// ---------------------------------------------------------------------------
__global__ void write_onehot(const int* __restrict__ eidx,
                             const int* __restrict__ keep,
                             float* __restrict__ out) {
    const int i = blockIdx.x * blockDim.x + threadIdx.x;   // 0 .. MTOK*NEXP-1
    if (i >= MTOK * NEXP) return;
    const int m = i >> 6;
    const int e = i & 63;
    out[i] = (eidx[m] == e && keep[m]) ? 1.0f : 0.0f;
}

// ---------------------------------------------------------------------------
// Kernel 3: pi reduction + aux loss + num_dropped. One block, 512 threads.
// ---------------------------------------------------------------------------
__global__ __launch_bounds__(512) void finalize(
    const float* __restrict__ probs,      // [MTOK, NEXP]
    const int*   __restrict__ kept_cnt,   // [BATCH*NEXP]
    const int*   __restrict__ drop_cnt,   // [BATCH*NEXP]
    float* __restrict__ aux_out,
    float* __restrict__ ndrop_out)
{
    __shared__ float redf[512];
    __shared__ int   redi[512];
    const int t = threadIdx.x;
    const int b = t >> 6;
    const int e = t & 63;

    const float* pp = probs + (size_t)b * SEQ * NEXP + e;
    float sum = 0.f;
    for (int s = 0; s < SEQ; ++s) sum += pp[(size_t)s * NEXP];
    const float pi = sum / (float)SEQ;
    const float fi = (float)kept_cnt[t] / (float)SEQ;
    redf[t] = fi * pi;
    redi[t] = drop_cnt[t];
    __syncthreads();
    for (int off = 256; off > 0; off >>= 1) {
        if (t < off) { redf[t] += redf[t + off]; redi[t] += redi[t + off]; }
        __syncthreads();
    }
    if (t == 0) {
        aux_out[0]   = redf[0] * (float)NEXP / (float)BATCH;
        ndrop_out[0] = (float)redi[0];
    }
}

// ---------------------------------------------------------------------------
// Launch
// ---------------------------------------------------------------------------
extern "C" void kernel_launch(void* const* d_in, const int* in_sizes, int n_in,
                              void* d_out, int out_size, void* d_ws, size_t ws_size,
                              hipStream_t stream) {
    const float* X    = (const float*)d_in[0];   // [B,S,D]
    const float* W    = (const float*)d_in[1];   // [D,E]
    const float* bias = (const float*)d_in[2];   // [E]

    float* out      = (float*)d_out;
    float* exp_ind  = out;                               // [B,S,E]  1,048,576
    float* top1     = out + (size_t)MTOK * NEXP;         // [B,S]       16,384
    float* probs    = top1 + MTOK;                       // [B,S,E]  1,048,576
    float* aux      = probs + (size_t)MTOK * NEXP;       // scalar
    float* ndrop    = aux + 1;                           // scalar

    char* ws = (char*)d_ws;
    __bf16* Wb     = (__bf16*)ws;                        // 512 KB (swizzled bf16 W)
    int* eidx      = (int*)(ws + (1u << 20));            // 64 KB
    int* keep      = (int*)(ws + (1u << 20) + (1u << 16));
    int* kept_cnt  = (int*)(ws + (1u << 20) + (2u << 16));
    int* drop_cnt  = kept_cnt + BATCH * NEXP;

    pack_w<<<(DIM * NEXP + 255) / 256, 256, 0, stream>>>(W, Wb);
    router_gemm_softmax<<<MTOK / 32, 32, 0, stream>>>(X, Wb, bias, probs, top1, eidx);
    capacity_scan<<<(BATCH * NEXP) / 8, 256, 0, stream>>>(eidx, keep, kept_cnt, drop_cnt);
    write_onehot<<<(MTOK * NEXP + 255) / 256, 256, 0, stream>>>(eidx, keep, exp_ind);
    finalize<<<1, 512, 0, stream>>>(probs, kept_cnt, drop_cnt, aux, ndrop);
}
